// VectorQuantizer_1340029796652
// MI455X (gfx1250) — matrix operand order repaired and checked by reading, not compiled, not run on hardware
//
#include <hip/hip_runtime.h>
#include <math.h>

// ---- CDNA5 WMMA / vector types ----
typedef __attribute__((ext_vector_type(16))) __bf16 v16bf;
typedef __attribute__((ext_vector_type(8)))  float  v8f;
typedef __attribute__((ext_vector_type(4)))  float  f4v;
typedef __attribute__((ext_vector_type(2)))  float  f2v;

#define N_ROWS   32768            // 32 * 32 * 32  (B*H*W)
#define E_DIM    256
#define N_E      1024
#define HW       1024             // H*W per batch
#define CHW      (E_DIM * HW)     // 262144 elements per batch of z
#define MTILE    128
#define BPAD     280              // padded LDS row (ushorts): 560B stride, 16B-aligned,
                                  // bank step 140 mod 64 = 12 -> conflict-free gathers

// d_out layout (float elements), tuple order: loss, z_q, perplexity, one_hot, indices
#define LOSS_OFF 0ull
#define ZQ_OFF   1ull
#define PPL_OFF  (1ull + 8388608ull)          // 8388609
#define OH_OFF   (PPL_OFF + 1ull)             // 8388610  (8B-aligned only!)
#define IDX_OFF  (OH_OFF + 33554432ull)       // 41943042

__device__ __forceinline__ unsigned int f2bf(float f) {
  unsigned int u = __float_as_uint(f);
  u += 0x7FFFu + ((u >> 16) & 1u);            // round-to-nearest-even
  return u >> 16;
}

// ---------------- Kernel 1: codebook norms + zero histogram ----------------
__global__ __launch_bounds__(32) void vq_prep(const float* __restrict__ cb,
                                              float* __restrict__ cnorms,
                                              int* __restrict__ hist) {
  const int k = blockIdx.x;
  const int lane = threadIdx.x;
  const float* row = cb + (size_t)k * E_DIM;
  float s = 0.0f;
#pragma unroll
  for (int i = 0; i < 8; ++i) {
    float v = row[lane + i * 32];
    s += v * v;
  }
#pragma unroll
  for (int off = 16; off > 0; off >>= 1) s += __shfl_xor(s, off, 32);
  if (lane == 0) {
    cnorms[k] = s;
    hist[k] = 0;
  }
}

// ------------- Kernel 2: bf16 WMMA distance GEMM + fused argmin -------------
// grid = N_ROWS/MTILE blocks, 256 threads (8 waves); wave w owns 16 rows.
// LDS B tile is double-buffered: tile t+1 is fetched from global while the
// 8 chunk WMMAs for tile t execute; one barrier per tile.
__global__ __launch_bounds__(256) void vq_gemm_argmin(
    const float* __restrict__ z, const float* __restrict__ cb,
    const float* __restrict__ cnorms, int* __restrict__ argmin_out) {
  __shared__ unsigned short ldsB[2][16 * BPAD];

  const int tid  = threadIdx.x;
  const int wave = tid >> 5;
  const int lane = tid & 31;
  const int m16  = lane & 15;                 // row-in-strip / code-in-tile
  const int half = lane >> 4;

  const int nw = blockIdx.x * MTILE + wave * 16;   // first row of this wave
  const int b  = nw >> 10;
  const int s0 = nw & 1023;
  const float* zb = z + (size_t)b * CHW;

  // staging helpers (256 threads cover the 16x256 f32 tile as 1024 float4)
  const int stCode = tid >> 6;                // base code row for this thread
  const int stDq   = tid & 63;                // float4 index within row

  // ---- A fragments: 16 rows x 256 dims, bf16, resident in registers ----
  union Frag { v16bf v; unsigned int u[8]; };
  Frag afrag[8];
#pragma unroll
  for (int kc = 0; kc < 8; ++kc) {
#pragma unroll
    for (int v = 0; v < 8; ++v) {
      // 16-bit A 16x32 layout: v<4 -> K = half*8 + 2v ; v>=4 -> K = 16 + half*8 + 2(v-4)
      const int kk = (v < 4) ? (half * 8 + 2 * v) : (16 + half * 8 + 2 * (v - 4));
      const int k  = kc * 32 + kk;
      const float f0 = zb[(size_t)k * HW + s0 + m16];
      const float f1 = zb[(size_t)(k + 1) * HW + s0 + m16];
      afrag[kc].u[v] = f2bf(f0) | (f2bf(f1) << 16);
    }
  }

  float minv[8];
  int   mini[8];
#pragma unroll
  for (int j = 0; j < 8; ++j) { minv[j] = 3.4e38f; mini[j] = 0; }

  // prologue: fetch + stage tile 0 into buffer 0
  float4 pre[4];
#pragma unroll
  for (int it = 0; it < 4; ++it) {
    const int code = stCode + it * 4;
    pre[it] = ((const float4*)(cb + (size_t)code * E_DIM))[stDq];
  }
#pragma unroll
  for (int it = 0; it < 4; ++it) {
    const int code = stCode + it * 4;
    unsigned int* p = (unsigned int*)&ldsB[0][code * BPAD + stDq * 4];
    p[0] = f2bf(pre[it].x) | (f2bf(pre[it].y) << 16);
    p[1] = f2bf(pre[it].z) | (f2bf(pre[it].w) << 16);
  }
  __syncthreads();

  for (int t = 0; t < N_E / 16; ++t) {        // 64 column tiles of 16 codes
    const int cur = t & 1;

    // fetch tile t+1 early so global latency overlaps the WMMAs below
    float4 nxt[4];
    if (t < N_E / 16 - 1) {
#pragma unroll
      for (int it = 0; it < 4; ++it) {
        const int code = (t + 1) * 16 + stCode + it * 4;
        nxt[it] = ((const float4*)(cb + (size_t)code * E_DIM))[stDq];
      }
    }

    v8f acc = {};
#pragma unroll
    for (int kc = 0; kc < 8; ++kc) {
      Frag bfrag;
#pragma unroll
      for (int v = 0; v < 8; ++v) {
        // 16-bit B 32x16 layout (matches A): v<4 -> K = half*8 + 2v ; v>=4 -> K = 16 + half*8 + 2(v-4)
        bfrag.u[v] =
            *(const unsigned int*)&ldsB[cur][m16 * BPAD + kc * 32 + ((v < 4) ? (half * 8 + 2 * v) : (16 + half * 8 + 2 * (v - 4)))];
      }
      acc = __builtin_amdgcn_wmma_f32_16x16x32_bf16(
          false, afrag[kc].v, false, bfrag.v, (short)0, acc, false, false);
    }

    const int   code = t * 16 + m16;          // this lane's code column
    const float cn   = cnorms[code];
#pragma unroll
    for (int j = 0; j < 8; ++j) {             // C/D layout: row = half*8 + j, col = m16
      const float dist = cn - 2.0f * acc[j];  // ||e||^2 - 2 z.e  (||z||^2 constant)
      if (dist < minv[j]) { minv[j] = dist; mini[j] = code; }
    }

    if (t < N_E / 16 - 1) {
      // buf[(t+1)&1] was last read in iteration t-1, before that iteration's
      // barrier -> safe to overwrite now; single barrier per tile.
#pragma unroll
      for (int it = 0; it < 4; ++it) {
        const int code2 = stCode + it * 4;
        unsigned int* p = (unsigned int*)&ldsB[cur ^ 1][code2 * BPAD + stDq * 4];
        p[0] = f2bf(nxt[it].x) | (f2bf(nxt[it].y) << 16);
        p[1] = f2bf(nxt[it].z) | (f2bf(nxt[it].w) << 16);
      }
      __syncthreads();
    }
  }

  // Cross-lane argmin over the 16 lanes sharing a row (groups of width 16)
#pragma unroll
  for (int j = 0; j < 8; ++j) {
    float v = minv[j];
    int   i = mini[j];
#pragma unroll
    for (int off = 8; off > 0; off >>= 1) {
      const float ov = __shfl_xor(v, off, 16);
      const int   oi = __shfl_xor(i, off, 16);
      if (ov < v) { v = ov; i = oi; }
    }
    if (m16 == 0) argmin_out[nw + half * 8 + j] = i;
  }
}

// ---- Kernel 3: indices, histogram, one-hot, gathered z_q, loss partials ----
// Streaming outputs use non-temporal stores so the 166MB of writes do not
// evict the L2-resident codebook; z is NT-loaded (last use).
__global__ __launch_bounds__(256) void vq_outputs(
    const float* __restrict__ z, const float* __restrict__ cb,
    const int* __restrict__ argmin, float* __restrict__ out,
    int* __restrict__ hist, float* __restrict__ lossPartial) {
  __shared__ int   sIdx[256];
  __shared__ float sRed[256];
  const int tid = threadIdx.x;
  const int n0  = blockIdx.x * 256;
  const int b   = n0 >> 10;
  const int s0  = n0 & 1023;

  const int myIdx = argmin[n0 + tid];
  sIdx[tid] = myIdx;
  out[IDX_OFF + (size_t)(n0 + tid)] = (float)myIdx;
  atomicAdd(&hist[myIdx], 1);

  const f4v* crow4 = (const f4v*)(cb + (size_t)myIdx * E_DIM);
  __builtin_prefetch(crow4, 0, 3);            // global_prefetch of gathered row
  __syncthreads();

  // One-hot rows: two coalesced NT float2 stores per row per thread
  // (output base is 8B-aligned, so float2 is the widest legal vector store)
  const int k0 = tid * 2;
  for (int r = 0; r < 256; ++r) {
    const int code = sIdx[r];
    float* row = out + OH_OFF + (size_t)(n0 + r) * N_E;
    f2v a, bb;
    a.x  = (k0       == code) ? 1.0f : 0.0f;
    a.y  = (k0 + 1   == code) ? 1.0f : 0.0f;
    bb.x = (k0 + 512 == code) ? 1.0f : 0.0f;
    bb.y = (k0 + 513 == code) ? 1.0f : 0.0f;
    __builtin_nontemporal_store(a,  (f2v*)row + tid);
    __builtin_nontemporal_store(bb, (f2v*)(row + 512) + tid);
  }

  // z_q gather (B,C,H,W): thread <-> spatial so stores are contiguous;
  // codebook row read as float4 (16B/lane per L2 line instead of 4B)
  const float* zbp = z + (size_t)b * CHW + s0 + tid;
  float*       zqp = out + ZQ_OFF + (size_t)b * CHW + s0 + tid;
  float acc = 0.0f;
#pragma unroll 4
  for (int cq = 0; cq < E_DIM / 4; ++cq) {
    const f4v q = crow4[cq];
#pragma unroll
    for (int i = 0; i < 4; ++i) {
      const size_t off = (size_t)(cq * 4 + i) * HW;
      const float zc = __builtin_nontemporal_load(zbp + off);
      __builtin_nontemporal_store(q[i], zqp + off);
      const float d = q[i] - zc;
      acc += d * d;
    }
  }
  sRed[tid] = acc;
  __syncthreads();
  for (int off = 128; off > 0; off >>= 1) {
    if (tid < off) sRed[tid] += sRed[tid + off];
    __syncthreads();
  }
  if (tid == 0) lossPartial[blockIdx.x] = sRed[0];   // deterministic partials
}

// ---------------- Kernel 4: loss + perplexity scalars ----------------
__global__ __launch_bounds__(256) void vq_finalize(
    const int* __restrict__ hist, const float* __restrict__ lossPartial,
    float* __restrict__ out) {
  __shared__ float sA[256];
  __shared__ float sB[256];
  const int tid = threadIdx.x;
  float h = 0.0f;
  for (int k = tid; k < N_E; k += 256) {
    const float e = (float)hist[k] * (1.0f / 32768.0f);
    h += e * logf(e + 1e-10f);
  }
  sA[tid] = h;
  sB[tid] = (tid < 128) ? lossPartial[tid] : 0.0f;
  __syncthreads();
  for (int off = 128; off > 0; off >>= 1) {
    if (tid < off) { sA[tid] += sA[tid + off]; sB[tid] += sB[tid + off]; }
    __syncthreads();
  }
  if (tid == 0) {
    out[PPL_OFF]  = expf(-sA[0]);
    out[LOSS_OFF] = sB[0] * 1.25f / 8388608.0f;  // (1+beta) * mean((z_q - z)^2)
  }
}

extern "C" void kernel_launch(void* const* d_in, const int* in_sizes, int n_in,
                              void* d_out, int out_size, void* d_ws, size_t ws_size,
                              hipStream_t stream) {
  const float* z  = (const float*)d_in[0];   // (32, 256, 32, 32) f32
  const float* cb = (const float*)d_in[1];   // (1024, 256) f32
  float* out = (float*)d_out;

  char* ws = (char*)d_ws;
  float* cnorms = (float*)(ws);              // 1024 f32       @ 0
  int*   hist   = (int*)(ws + 4096);         // 1024 i32       @ 4096
  float* lossP  = (float*)(ws + 8192);       // 128 f32        @ 8192
  int*   argmin = (int*)(ws + 9216);         // 32768 i32      @ 9216

  vq_prep<<<N_E, 32, 0, stream>>>(cb, cnorms, hist);
  vq_gemm_argmin<<<N_ROWS / MTILE, 256, 0, stream>>>(z, cb, cnorms, argmin);
  vq_outputs<<<N_ROWS / 256, 256, 0, stream>>>(z, cb, argmin, out, hist, lossP);
  vq_finalize<<<1, 256, 0, stream>>>(hist, lossP, out);
}
